// MyNewGNN_76476187673066
// MI455X (gfx1250) — compile-verified, matrix operand order checked by reading, not run on hardware
//
#include <hip/hip_runtime.h>
#include <hip/hip_bf16.h>
#include <math.h>

// ---------------- model constants ----------------
#define NCLASS 256
#define NFEAT  128
#define BATCH  32
#define NNODE  512
#define NB     (BATCH * NNODE)   // 16384 nodes
#define EDGES  (NB * 16)         // 262144 edges
#define RNNIN  300
#define T_SU   100
#define T_SV   80

typedef __attribute__((ext_vector_type(16))) _Float16 v16h;
typedef __attribute__((ext_vector_type(8)))  float    v8f;

// ===================================================================
// f32 -> f16 conversion with K zero-padding to a multiple of 32.
// Activations/weights are converted once so the GEMM can DMA pure
// f16 tiles with CDNA5 async global->LDS copies (no K-edge guards).
// ===================================================================
__global__ void cvt_f16_kernel(const float* __restrict__ src, _Float16* __restrict__ dst,
                               int M, int K, int Kp)
{
    int i = blockIdx.x * blockDim.x + threadIdx.x;
    if (i >= M * Kp) return;
    int m = i / Kp, k = i - m * Kp;
    dst[i] = (k < K) ? (_Float16)src[(size_t)m * K + k] : (_Float16)0.0f;
}

// ===================================================================
// Tiled WMMA GEMM:  Y[M x N] = X16[M x Kp] * W16[N x Kp]^T (+bias)(+Cin)(ReLU?)
// 256 threads = 8 waves; tile 128x128; K stepped by 32.
// Double-buffered LDS staged with GLOBAL_LOAD_ASYNC_TO_LDS_B128
// (ASYNCcnt / s_wait_asynccnt, ISA 08_async_tensor), overlapping the
// DMA of tile k+1 with the WMMA chain of tile k.
// ===================================================================
#define TM 128
#define TN 128
#define TK 32
#define LDSK 40   // padded LDS row stride in halves (80 B, 16B-aligned rows)

__device__ __forceinline__ void async_tile_b128(unsigned lds_off, const _Float16* gptr)
{
    asm volatile("global_load_async_to_lds_b128 %0, %1, off"
                 :: "v"(lds_off), "v"(gptr) : "memory");
}

__device__ __forceinline__ void wait_async0()
{
    asm volatile("s_wait_asynccnt 0" ::: "memory");
}

// stage one 128x32 f16 tile (rows row0.., k offset k0) into LDS tile base
__device__ __forceinline__ void stage_tile(const _Float16* __restrict__ src, int rows_total,
                                           int Kp, int row0, int k0, _Float16* lds)
{
    const int tid = threadIdx.x;
    #pragma unroll
    for (int h = 0; h < 2; ++h) {            // 512 16-byte chunks / 256 threads
        int c  = tid + h * 256;
        int r  = c >> 2;                      // 4 chunks (32 halves) per row
        int kk = (c & 3) * 8;
        _Float16* ldst = lds + r * LDSK + kk;
        int gr = row0 + r;
        if (gr < rows_total) {
            const _Float16* g = src + (size_t)gr * Kp + (k0 + kk);
            async_tile_b128((unsigned)(uintptr_t)ldst, g);
        } else {
            int4 zz; zz.x = 0; zz.y = 0; zz.z = 0; zz.w = 0;   // zero-fill OOB rows
            *reinterpret_cast<int4*>(ldst) = zz;
        }
    }
}

__global__ __launch_bounds__(256, 2)
void wmma_gemm16_kernel(const _Float16* __restrict__ X, const _Float16* __restrict__ W,
                        const float* __restrict__ bias, const float* Cin,
                        float* Y, int M, int N, int K /*padded*/, int relu)
{
    __shared__ _Float16 As[2][TM * LDSK];
    __shared__ _Float16 Bs[2][TN * LDSK];

    const int tid   = threadIdx.x;
    const int lane  = tid & 31;
    const int wv    = tid >> 5;              // wave 0..7
    const int m0    = blockIdx.x * TM;
    const int n0    = blockIdx.y * TN;
    const int rbase = wv * 16;               // wave's 16-row band
    const int colq  = lane & 15;
    const int rowoff = (lane < 16) ? 0 : 8;  // C layout: lanes>=16 hold M=r+8
    const int kq     = (lane < 16) ? 0 : 8;  // A/B K split per half-wave

    // --- init accumulators from bias / Cin (C layout per ISA 7.12.2) ---
    v8f acc[8];
    for (int ns = 0; ns < 8; ++ns) {
        int col = n0 + ns * 16 + colq;
        float bval = (bias != nullptr && col < N) ? bias[col] : 0.0f;
        for (int r = 0; r < 8; ++r) {
            int row = m0 + rbase + rowoff + r;
            float cv = 0.0f;
            if (Cin != nullptr && row < M && col < N)
                cv = Cin[(size_t)row * N + col];
            acc[ns][r] = cv + bval;
        }
    }

    const int nk = K / TK;
    // prologue: stage tile 0 into buffer 0
    stage_tile(X, M, K, m0, 0, As[0]);
    stage_tile(W, N, K, n0, 0, Bs[0]);
    wait_async0();
    __syncthreads();

    for (int kt = 0; kt < nk; ++kt) {
        const int cur = kt & 1;
        // kick off DMA for next tile while we compute on the current one
        if (kt + 1 < nk) {
            stage_tile(X, M, K, m0, (kt + 1) * TK, As[cur ^ 1]);
            stage_tile(W, N, K, n0, (kt + 1) * TK, Bs[cur ^ 1]);
        }

        // A fragment: lane holds row (lane&15); K pattern kq..kq+7, 16+kq..16+kq+7
        v16h afrag;
        {
            const _Float16* ap = &As[cur][(rbase + colq) * LDSK];
            #pragma unroll
            for (int j = 0; j < 8; ++j) afrag[j]     = ap[kq + j];
            #pragma unroll
            for (int j = 0; j < 8; ++j) afrag[8 + j] = ap[16 + kq + j];
        }
        #pragma unroll
        for (int ns = 0; ns < 8; ++ns) {
            v16h bfrag;   // B = W^T: lane holds output column (stored row of W tile)
            const _Float16* bp = &Bs[cur][(ns * 16 + colq) * LDSK];
            #pragma unroll
            for (int j = 0; j < 8; ++j) bfrag[j]     = bp[kq + j];
            #pragma unroll
            for (int j = 0; j < 8; ++j) bfrag[8 + j] = bp[16 + kq + j];
            acc[ns] = __builtin_amdgcn_wmma_f32_16x16x32_f16(
                false, afrag, false, bfrag, (short)0, acc[ns], false, false);
        }

        wait_async0();      // next tile DMA complete (per-wave)
        __syncthreads();    // all waves' chunks visible before next iteration
    }

    for (int ns = 0; ns < 8; ++ns) {
        int col = n0 + ns * 16 + colq;
        if (col >= N) continue;
        for (int r = 0; r < 8; ++r) {
            int row = m0 + rbase + rowoff + r;
            if (row >= M) continue;
            float v = acc[ns][r];
            if (relu) v = fmaxf(v, 0.0f);
            Y[(size_t)row * N + col] = v;
        }
    }
}

// ===================================================================
// Edge aggregation (SAGE mean): scatter-add + degree + normalize
// ===================================================================
__global__ void fill_zero_kernel(float* p, int n) {
    int i = blockIdx.x * blockDim.x + threadIdx.x;
    if (i < n) p[i] = 0.0f;
}

__global__ void scatter_deg_kernel(const int* __restrict__ ei, float* deg, int E) {
    int e = blockIdx.x * blockDim.x + threadIdx.x;
    if (e < E) atomicAdd(&deg[ei[E + e]], 1.0f);
}

__global__ void scatter_feat_kernel(const int* __restrict__ ei,
                                    const float* __restrict__ x, float* s, int E) {
    int t = blockIdx.x * blockDim.x + threadIdx.x;       // (edge, 8-float chunk)
    if (t >= E * (NCLASS / 8)) return;
    int e = t >> 5;             // NCLASS/8 == 32 chunks
    int c = t & 31;
    int sn = ei[e], dn = ei[E + e];
    const float* xs = x + (size_t)sn * NCLASS + c * 8;
    float*       sd = s + (size_t)dn * NCLASS + c * 8;
    #pragma unroll
    for (int j = 0; j < 8; ++j) atomicAdd(&sd[j], xs[j]);
}

__global__ void normalize_kernel(const float* __restrict__ deg, float* s, int n) {
    int i = blockIdx.x * blockDim.x + threadIdx.x;
    if (i >= n * NCLASS) return;
    float d = deg[i >> 8];
    s[i] = s[i] / fmaxf(d, 1.0f);
}

// ===================================================================
// GRU recurrence only (input projection gi precomputed on WMMA path).
// One launch, two blocks: block 0 = forward, block 1 = backward.
// ===================================================================
__global__ __launch_bounds__(256)
void gru_rec_kernel(const float* __restrict__ giF, const float* __restrict__ giB, int T,
                    const float* __restrict__ whhF, const float* __restrict__ bhhF,
                    const float* __restrict__ whhB, const float* __restrict__ bhhB,
                    float* __restrict__ yF, float* __restrict__ yB)
{
    const int dir = blockIdx.x;
    const float* gi  = dir ? giB : giF;
    const float* whh = dir ? whhB : whhF;
    const float* bhh = dir ? bhhB : bhhF;
    float* y = dir ? yB : yF;

    __shared__ float h[NCLASS];
    const int t = threadIdx.x;
    h[t] = 0.0f;
    __syncthreads();

    const float* ur_ = whh + (size_t)t * NCLASS;
    const float* uz_ = whh + (size_t)(NCLASS + t) * NCLASS;
    const float* un_ = whh + (size_t)(2 * NCLASS + t) * NCLASS;
    const float br = bhh[t], bz = bhh[NCLASS + t], bn = bhh[2 * NCLASS + t];

    for (int s = 0; s < T; ++s) {
        int ts = dir ? (T - 1 - s) : s;
        const float* g = gi + (size_t)ts * (3 * NCLASS);
        float hr = br, hz = bz, hn = bn;
        for (int k = 0; k < NCLASS; ++k) {
            float hv = h[k];
            hr += ur_[k] * hv; hz += uz_[k] * hv; hn += un_[k] * hv;
        }
        float r = 1.0f / (1.0f + __expf(-(g[t] + hr)));
        float z = 1.0f / (1.0f + __expf(-(g[NCLASS + t] + hz)));
        float n = tanhf(g[2 * NCLASS + t] + r * hn);
        float hnew = (1.0f - z) * n + z * h[t];
        __syncthreads();
        h[t] = hnew;
        y[(size_t)ts * NCLASS + t] = hnew;
        __syncthreads();
    }
}

__global__ void rnn_reduce_kernel(const float* __restrict__ yf,
                                  const float* __restrict__ yb, int T, float* out)
{
    int f = blockIdx.x * blockDim.x + threadIdx.x;
    if (f >= 2 * NCLASS) return;
    const float* src = (f < NCLASS) ? yf : yb;
    int c = (f < NCLASS) ? f : f - NCLASS;
    float acc = 0.0f;
    for (int s = 0; s < T; ++s) acc += fmaxf(src[(size_t)s * NCLASS + c], 0.0f);
    out[f] = acc / (float)T;
}

// ===================================================================
// Set2Set pooling: one workgroup per graph (32 graphs x 512 nodes, d=256)
// ===================================================================
__global__ __launch_bounds__(256)
void set2set_kernel(const float* __restrict__ x, const float* __restrict__ wih,
                    const float* __restrict__ whh, const float* __restrict__ bih,
                    const float* __restrict__ bhh, float* __restrict__ qout)
{
    __shared__ float h[NCLASS], c[NCLASS], q[2 * NCLASS], e[NNODE], red[256];
    const int t = threadIdx.x;
    const int base = blockIdx.x * NNODE;
    h[t] = 0.0f; c[t] = 0.0f; q[t] = 0.0f; q[NCLASS + t] = 0.0f;
    __syncthreads();

    for (int step = 0; step < 2; ++step) {
        float gi = bih[t] + bhh[t];
        float gf = bih[NCLASS + t] + bhh[NCLASS + t];
        float gg = bih[2 * NCLASS + t] + bhh[2 * NCLASS + t];
        float go = bih[3 * NCLASS + t] + bhh[3 * NCLASS + t];
        for (int k = 0; k < 2 * NCLASS; ++k) {
            float qv = q[k];
            gi += wih[(size_t)t * (2 * NCLASS) + k] * qv;
            gf += wih[(size_t)(NCLASS + t) * (2 * NCLASS) + k] * qv;
            gg += wih[(size_t)(2 * NCLASS + t) * (2 * NCLASS) + k] * qv;
            go += wih[(size_t)(3 * NCLASS + t) * (2 * NCLASS) + k] * qv;
        }
        for (int k = 0; k < NCLASS; ++k) {
            float hv = h[k];
            gi += whh[(size_t)t * NCLASS + k] * hv;
            gf += whh[(size_t)(NCLASS + t) * NCLASS + k] * hv;
            gg += whh[(size_t)(2 * NCLASS + t) * NCLASS + k] * hv;
            go += whh[(size_t)(3 * NCLASS + t) * NCLASS + k] * hv;
        }
        float si = 1.0f / (1.0f + __expf(-gi));
        float sf = 1.0f / (1.0f + __expf(-gf));
        float so = 1.0f / (1.0f + __expf(-go));
        float cn = sf * c[t] + si * tanhf(gg);
        float hn = so * tanhf(cn);
        __syncthreads();
        c[t] = cn; h[t] = hn;
        __syncthreads();

        float lmax = -3.4e38f;
        for (int i = t; i < NNODE; i += 256) {
            const float* xr = x + (size_t)(base + i) * NCLASS;
            float d = 0.0f;
            for (int k = 0; k < NCLASS; ++k) d += xr[k] * h[k];
            e[i] = d;
            lmax = fmaxf(lmax, d);
        }
        red[t] = lmax; __syncthreads();
        for (int off = 128; off > 0; off >>= 1) {
            if (t < off) red[t] = fmaxf(red[t], red[t + off]);
            __syncthreads();
        }
        float emax = red[0]; __syncthreads();
        float lsum = 0.0f;
        for (int i = t; i < NNODE; i += 256) {
            float a = __expf(e[i] - emax); e[i] = a; lsum += a;
        }
        red[t] = lsum; __syncthreads();
        for (int off = 128; off > 0; off >>= 1) {
            if (t < off) red[t] += red[t + off];
            __syncthreads();
        }
        float inv = 1.0f / red[0]; __syncthreads();

        float rv = 0.0f;
        for (int i = 0; i < NNODE; ++i)
            rv += e[i] * x[(size_t)(base + i) * NCLASS + t];
        rv *= inv;
        q[t] = h[t]; q[NCLASS + t] = rv;
        __syncthreads();
    }
    qout[(size_t)blockIdx.x * (2 * NCLASS) + t]          = q[t];
    qout[(size_t)blockIdx.x * (2 * NCLASS) + NCLASS + t] = q[NCLASS + t];
}

// ===================================================================
// Head input concat: D[32][2048] = [su | su_sm | sv | sv_sm]
// ===================================================================
__global__ void build_head_in_kernel(const float* __restrict__ su, const float* __restrict__ susm,
                                     const float* __restrict__ sv, const float* __restrict__ svsm,
                                     float* D)
{
    int idx = blockIdx.x * blockDim.x + threadIdx.x;
    if (idx >= BATCH * 2048) return;
    int b = idx >> 11, j = idx & 2047;
    float v;
    if      (j < 512)  v = su[b * 512 + j];
    else if (j < 1024) v = susm[j - 512];
    else if (j < 1536) v = sv[b * 512 + (j - 1024)];
    else               v = svsm[j - 1536];
    D[idx] = v;
}

// ===================================================================
// Host-side orchestration
// ===================================================================
enum {
    FC1_W = 0, FC1_B,
    SU1_WL, SU1_BL, SU1_WR, SU2_WL, SU2_BL, SU2_WR,
    SV1_WL, SV1_BL, SV1_WR, SV2_WL, SV2_BL, SV2_WR,
    GRUF_WIH, GRUF_WHH, GRUF_BIH, GRUF_BHH,
    GRUB_WIH, GRUB_WHH, GRUB_BIH, GRUB_BHH,
    S2S_WIH, S2S_WHH, S2S_BIH, S2S_BHH,
    FC2_W, FC2_B, FC3_W, FC3_B, FC4_W, FC4_B, FC5_W, FC5_B,
    NPARAMS
};

// convert X,W to padded f16 then run the async-DMA WMMA GEMM
static void gemm16(hipStream_t st, const float* X, const float* W, const float* bias,
                   const float* Cin, float* Y, int M, int N, int K, int relu,
                   _Float16* x16, _Float16* w16)
{
    int Kp = (K + 31) & ~31;
    cvt_f16_kernel<<<(M * Kp + 255) / 256, 256, 0, st>>>(X, x16, M, K, Kp);
    cvt_f16_kernel<<<(N * Kp + 255) / 256, 256, 0, st>>>(W, w16, N, K, Kp);
    dim3 grid((M + TM - 1) / TM, (N + TN - 1) / TN);
    wmma_gemm16_kernel<<<grid, 256, 0, st>>>(x16, w16, bias, Cin, Y, M, N, Kp, relu);
}

static void run_sage(hipStream_t st, const float* xin, const int* ei,
                     const float* wl, const float* bl, const float* wr,
                     const float* resid, float* s, float* deg, float* Y, int relu,
                     _Float16* x16, _Float16* w16)
{
    const int nbf = NB * NCLASS;
    fill_zero_kernel<<<(nbf + 255) / 256, 256, 0, st>>>(s, nbf);
    fill_zero_kernel<<<(NB + 255) / 256, 256, 0, st>>>(deg, NB);
    scatter_deg_kernel<<<(EDGES + 255) / 256, 256, 0, st>>>(ei, deg, EDGES);
    scatter_feat_kernel<<<(EDGES * 32 + 255) / 256, 256, 0, st>>>(ei, xin, s, EDGES);
    normalize_kernel<<<(nbf + 255) / 256, 256, 0, st>>>(deg, s, NB);
    // Y = neigh @ wl^T + bl (+resid)   then   Y += xin @ wr^T  (fused ReLU opt)
    gemm16(st, s, wl, bl, resid, Y, NB, NCLASS, NCLASS, 0, x16, w16);
    gemm16(st, xin, wr, nullptr, Y, Y, NB, NCLASS, NCLASS, relu, x16, w16);
}

static void run_block(hipStream_t st, const float* x128, const int* ei,
                      const float* const* P, int wl1i,
                      float* init, float* h, float* s, float* deg, float* out2,
                      _Float16* x16, _Float16* w16)
{
    gemm16(st, x128, P[FC1_W], P[FC1_B], nullptr, init, NB, NCLASS, NFEAT, 0, x16, w16);
    run_sage(st, init, ei, P[wl1i], P[wl1i + 1], P[wl1i + 2], nullptr, s, deg, h, 1, x16, w16);
    run_sage(st, h,    ei, P[wl1i + 3], P[wl1i + 4], P[wl1i + 5], init, s, deg, out2, 0, x16, w16);
}

static void run_rnn(hipStream_t st, const float* xs, int T, const float* const* P,
                    float* gif, float* gib, float* yf, float* yb, float* out512,
                    _Float16* x16, _Float16* w16)
{
    // input projections for ALL timesteps on the WMMA path (K 300 -> 320 padded)
    gemm16(st, xs, P[GRUF_WIH], P[GRUF_BIH], nullptr, gif, T, 3 * NCLASS, RNNIN, 0, x16, w16);
    gemm16(st, xs, P[GRUB_WIH], P[GRUB_BIH], nullptr, gib, T, 3 * NCLASS, RNNIN, 0, x16, w16);
    // both directions concurrently, recurrence only
    gru_rec_kernel<<<2, 256, 0, st>>>(gif, gib, T, P[GRUF_WHH], P[GRUF_BHH],
                                      P[GRUB_WHH], P[GRUB_BHH], yf, yb);
    rnn_reduce_kernel<<<2, 256, 0, st>>>(yf, yb, T, out512);
}

extern "C" void kernel_launch(void* const* d_in, const int* in_sizes, int n_in,
                              void* d_out, int out_size, void* d_ws, size_t ws_size,
                              hipStream_t stream)
{
    (void)in_sizes; (void)n_in; (void)out_size; (void)ws_size;

    const float* solute_x[5];   for (int i = 0; i < 5; ++i) solute_x[i]  = (const float*)d_in[i];
    const float* solvent_x[5];  for (int i = 0; i < 5; ++i) solvent_x[i] = (const float*)d_in[5 + i];
    const float* su_smiles = (const float*)d_in[10];
    const float* sv_smiles[5];  for (int i = 0; i < 5; ++i) sv_smiles[i] = (const float*)d_in[11 + i];
    const int*   su_adj = (const int*)d_in[16];
    const int*   sv_adj[5];     for (int i = 0; i < 5; ++i) sv_adj[i]    = (const int*)d_in[17 + i];
    const float* P[NPARAMS];
    for (int i = 0; i < NPARAMS; ++i) P[i] = (const float*)d_in[22 + i];

    float* out = (float*)d_out;   // [5*BATCH, 1] = 160 floats

    // ---- workspace carve (float units) ----
    float* ws = (float*)d_ws;  size_t off = 0;
    float* init = ws + off; off += (size_t)NB * NCLASS;
    float* hbuf = ws + off; off += (size_t)NB * NCLASS;
    float* sbuf = ws + off; off += (size_t)NB * NCLASS;
    float* out2 = ws + off; off += (size_t)NB * NCLASS;
    float* deg  = ws + off; off += NB;
    float* su   = ws + off; off += BATCH * 512;
    float* sv   = ws + off; off += BATCH * 512;
    float* susm = ws + off; off += 512;
    float* svsm = ws + off; off += 512;
    float* yf   = ws + off; off += (size_t)T_SU * NCLASS;
    float* yb   = ws + off; off += (size_t)T_SU * NCLASS;
    float* gif  = ws + off; off += (size_t)T_SU * 3 * NCLASS;
    float* gib  = ws + off; off += (size_t)T_SU * 3 * NCLASS;
    float* Dbuf = ws + off; off += (size_t)BATCH * 2048;
    float* t1   = ws + off; off += BATCH * 256;
    float* t2   = ws + off; off += BATCH * 64;
    float* t3   = ws + off; off += BATCH * 32;
    // f16 staging buffers (carved in float units: 2 halves per float)
    _Float16* x16 = (_Float16*)(ws + off); off += (size_t)NB * NCLASS / 2;      // up to 16384x256 halves
    _Float16* w16 = (_Float16*)(ws + off); off += (size_t)(NCLASS * 2048) / 2;  // up to 256x2048 halves

    // solute SMILES embedding computed once
    run_rnn(stream, su_smiles, T_SU, P, gif, gib, yf, yb, susm, x16, w16);

    for (int it = 0; it < 5; ++it) {
        // solute GNN block + set2set
        run_block(stream, solute_x[it], su_adj, P, SU1_WL, init, hbuf, sbuf, deg, out2, x16, w16);
        set2set_kernel<<<BATCH, 256, 0, stream>>>(out2, P[S2S_WIH], P[S2S_WHH],
                                                  P[S2S_BIH], P[S2S_BHH], su);
        // solvent GNN block + set2set
        run_block(stream, solvent_x[it], sv_adj[it], P, SV1_WL, init, hbuf, sbuf, deg, out2, x16, w16);
        set2set_kernel<<<BATCH, 256, 0, stream>>>(out2, P[S2S_WIH], P[S2S_WHH],
                                                  P[S2S_BIH], P[S2S_BHH], sv);
        // solvent SMILES embedding
        run_rnn(stream, sv_smiles[it], T_SV, P, gif, gib, yf, yb, svsm, x16, w16);

        // head MLP (all on WMMA GEMM path)
        build_head_in_kernel<<<(BATCH * 2048 + 255) / 256, 256, 0, stream>>>(su, susm, sv, svsm, Dbuf);
        gemm16(stream, Dbuf, P[FC2_W], P[FC2_B], nullptr, t1, BATCH, 256, 2048, 1, x16, w16);
        gemm16(stream, t1,   P[FC3_W], P[FC3_B], nullptr, t2, BATCH, 64, 256, 1, x16, w16);
        gemm16(stream, t2,   P[FC4_W], P[FC4_B], nullptr, t3, BATCH, 32, 64, 1, x16, w16);
        gemm16(stream, t3,   P[FC5_W], P[FC5_B], nullptr, out + it * BATCH, BATCH, 1, 32, 0, x16, w16);
    }
}
